// GCNStack_30769145708863
// MI455X (gfx1250) — compile-verified, matrix operand order checked by reading, not run on hardware
//
#include <hip/hip_runtime.h>
#include <hip/hip_bf16.h>

typedef __attribute__((ext_vector_type(16))) _Float16 v16h;
typedef __attribute__((ext_vector_type(8)))  float    v8f;

#define HID 128

// ---------------------------------------------------------------------------
// Pre-pack weights f32 [K,128] into per-(ktile, coltile, lane) v16h fragments
// matching the CDNA5 16-bit B-matrix (32x16) layout (lane = column, K halves
// split 0..7 / 16..23 per half-wave, mirroring the documented A layout).
// One thread writes one lane's 16-half fragment (32 contiguous bytes).
// ---------------------------------------------------------------------------
__global__ void pack_w_kernel(const float* __restrict__ W, _Float16* __restrict__ Wp, int K) {
    int idx = blockIdx.x * blockDim.x + threadIdx.x;
    int kt = K >> 5;
    int total = kt * 8 * 32;              // ktiles x 8 col-tiles x 32 lanes
    if (idx >= total) return;
    int lane = idx & 31;
    int c    = (idx >> 5) & 7;
    int t    = idx >> 8;
    int col  = c * 16 + (lane & 15);
    int kb   = t * 32 + ((lane & 16) ? 8 : 0);
    _Float16* out = Wp + (size_t)idx * 16;
#pragma unroll
    for (int i = 0; i < 8; ++i) {
        out[i]     = (_Float16)W[(size_t)(kb + i)      * HID + col];
        out[8 + i] = (_Float16)W[(size_t)(kb + 16 + i) * HID + col];
    }
}

// ---------------------------------------------------------------------------
// WMMA GEMM: out[n,128] = A[n,K] @ W (prepacked f16 fragments), f32 accumulate.
// Block = 256 threads = 8 waves; wave w computes rows [16*bid,+16) x cols
// [16*w,+16). A fragments loaded straight from f32 memory (two 32B runs per
// lane per k-step) and converted in-register to f16.
// ---------------------------------------------------------------------------
__global__ void gemm_wmma_kernel(const float* __restrict__ A, const _Float16* __restrict__ Wp,
                                 float* __restrict__ out, int n, int K) {
    int wave = threadIdx.x >> 5;
    int lane = threadIdx.x & 31;
    int rowBase = blockIdx.x * 16;
    int row = rowBase + (lane & 15);
    if (row >= n) row = n - 1;            // clamp loads; stores predicated (EXEC all-1 for WMMA)
    int kHalf = (lane & 16) ? 8 : 0;
    v8f acc = {};
    int kt = K >> 5;
    for (int t = 0; t < kt; ++t) {
        int kb = t * 32 + kHalf;
        const float* ar = A + (size_t)row * K + kb;
        float4 a0 = *(const float4*)(ar);
        float4 a1 = *(const float4*)(ar + 4);
        float4 a2 = *(const float4*)(ar + 16);
        float4 a3 = *(const float4*)(ar + 20);
        v16h af;
        af[0]  = (_Float16)a0.x; af[1]  = (_Float16)a0.y; af[2]  = (_Float16)a0.z; af[3]  = (_Float16)a0.w;
        af[4]  = (_Float16)a1.x; af[5]  = (_Float16)a1.y; af[6]  = (_Float16)a1.z; af[7]  = (_Float16)a1.w;
        af[8]  = (_Float16)a2.x; af[9]  = (_Float16)a2.y; af[10] = (_Float16)a2.z; af[11] = (_Float16)a2.w;
        af[12] = (_Float16)a3.x; af[13] = (_Float16)a3.y; af[14] = (_Float16)a3.z; af[15] = (_Float16)a3.w;
        v16h bf = *(const v16h*)(Wp + ((size_t)(t * 8 + wave) * 32 + lane) * 16);
        acc = __builtin_amdgcn_wmma_f32_16x16x32_f16(false, af, false, bf, (short)0, acc,
                                                     false, false);
    }
    // D layout: VGPR j -> M = j + 8*(lane>=16), N = lane&15
    int col = wave * 16 + (lane & 15);
    int rb  = rowBase + ((lane & 16) ? 8 : 0);
#pragma unroll
    for (int j = 0; j < 8; ++j) {
        int r = rb + j;
        if (r < n) out[(size_t)r * HID + col] = acc[j];
    }
}

// ---------------------------------------------------------------------------
// Degree / norm precompute (self-loops folded in: deg starts at 1)
// ---------------------------------------------------------------------------
__global__ void deg_init_kernel(float* __restrict__ deg, int n) {
    int i = blockIdx.x * blockDim.x + threadIdx.x;
    if (i < n) deg[i] = 1.0f;
}

__global__ void deg_edge_kernel(const int* __restrict__ dst, float* __restrict__ deg, int E) {
    int e = blockIdx.x * blockDim.x + threadIdx.x;
    if (e < E) atomicAdd(&deg[dst[e]], 1.0f);
}

__global__ void dinv_kernel(const float* __restrict__ deg, float* __restrict__ dinv,
                            float* __restrict__ dinv2, int n) {
    int i = blockIdx.x * blockDim.x + threadIdx.x;
    if (i < n) {
        float r = rsqrtf(deg[i]);
        dinv[i]  = r;
        dinv2[i] = r * r;                 // self-loop norm = dinv[i]^2
    }
}

__global__ void norm_kernel(const int* __restrict__ src, const int* __restrict__ dst,
                            const float* __restrict__ dinv, float* __restrict__ nrm, int E) {
    int e = blockIdx.x * blockDim.x + threadIdx.x;
    if (e < E) nrm[e] = dinv[src[e]] * dinv[dst[e]];
}

// agg[i,:] = m[i,:] * dinv2[i] + b  (self-loop contribution + bias)
__global__ void agg_init_kernel(const float* __restrict__ m, const float* __restrict__ b,
                                const float* __restrict__ dinv2, float* __restrict__ agg, int n) {
    int idx = blockIdx.x * blockDim.x + threadIdx.x;
    if (idx >= n * 32) return;
    int i = idx >> 5;
    int c = (idx & 31) * 4;
    size_t base = (size_t)i * HID + c;
    float d2 = dinv2[i];
    float4 v = *(const float4*)(m + base);
    float4 o;
    o.x = v.x * d2 + b[c];     o.y = v.y * d2 + b[c + 1];
    o.z = v.z * d2 + b[c + 2]; o.w = v.w * d2 + b[c + 3];
    *(float4*)(agg + base) = o;
}

// one wave (32 lanes) per edge; lane covers 4 channels: float4 gather + 4 atomics
__global__ void edge_scatter_kernel(const int* __restrict__ src, const int* __restrict__ dst,
                                    const float* __restrict__ nrm, const float* __restrict__ m,
                                    float* __restrict__ agg, int E) {
    int gid = blockIdx.x * blockDim.x + threadIdx.x;
    int e = gid >> 5;
    if (e >= E) return;
    int lane = threadIdx.x & 31;
    int s = src[e], d = dst[e];
    float w = nrm[e];
    float4 v = *(const float4*)(m + (size_t)s * HID + lane * 4);
    float* ap = agg + (size_t)d * HID + lane * 4;
    atomicAdd(ap + 0, v.x * w);
    atomicAdd(ap + 1, v.y * w);
    atomicAdd(ap + 2, v.z * w);
    atomicAdd(ap + 3, v.w * w);
}

// one wave per node: LayerNorm over 128 ch (4 per lane) + ReLU + residual add
__global__ void ln_relu_res_kernel(const float* __restrict__ agg, const float* __restrict__ g,
                                   const float* __restrict__ be, const float* __restrict__ resid,
                                   float* __restrict__ out, int n) {
    int wid  = (blockIdx.x * blockDim.x + threadIdx.x) >> 5;
    int lane = threadIdx.x & 31;
    if (wid >= n) return;
    size_t base = (size_t)wid * HID + lane * 4;
    float4 v = *(const float4*)(agg + base);
    float s = v.x + v.y + v.z + v.w;
#pragma unroll
    for (int off = 16; off >= 1; off >>= 1) s += __shfl_xor(s, off, 32);
    float mu = s * (1.0f / HID);
    float dx = v.x - mu, dy = v.y - mu, dz = v.z - mu, dw = v.w - mu;
    float q = dx * dx + dy * dy + dz * dz + dw * dw;
#pragma unroll
    for (int off = 16; off >= 1; off >>= 1) q += __shfl_xor(q, off, 32);
    float r = rsqrtf(q * (1.0f / HID) + 1e-5f);
    int c = lane * 4;
    float4 res = *(const float4*)(resid + base);
    float4 o;
    o.x = fmaxf(dx * r * g[c]     + be[c],     0.0f) + res.x;
    o.y = fmaxf(dy * r * g[c + 1] + be[c + 1], 0.0f) + res.y;
    o.z = fmaxf(dz * r * g[c + 2] + be[c + 2], 0.0f) + res.z;
    o.w = fmaxf(dw * r * g[c + 3] + be[c + 3], 0.0f) + res.w;
    *(float4*)(out + base) = o;
}

// ---------------------------------------------------------------------------
extern "C" void kernel_launch(void* const* d_in, const int* in_sizes, int n_in,
                              void* d_out, int out_size, void* d_ws, size_t ws_size,
                              hipStream_t stream) {
    const float* x    = (const float*)d_in[0];
    const int*   ei   = (const int*)  d_in[1];
    const float* W0   = (const float*)d_in[2];
    const float* b0   = (const float*)d_in[3];
    const float* g0   = (const float*)d_in[4];
    const float* be0  = (const float*)d_in[5];
    const float* W1   = (const float*)d_in[6];
    const float* b1   = (const float*)d_in[7];
    const float* g1   = (const float*)d_in[8];
    const float* be1  = (const float*)d_in[9];
    const float* W2   = (const float*)d_in[10];
    const float* b2   = (const float*)d_in[11];
    const float* g2   = (const float*)d_in[12];
    const float* be2  = (const float*)d_in[13];
    const float* resW = (const float*)d_in[14];

    const int IN_CH = 256;
    int N = in_sizes[0] / IN_CH;
    int E = in_sizes[1] / 2;
    const int* src = ei;
    const int* dst = ei + E;
    float* h = (float*)d_out;             // running node features [N,128]

    // bump allocator over d_ws (256B aligned)
    char* p = (char*)d_ws;
    auto alloc = [&](size_t bytes) -> char* {
        char* r = p;
        p += (bytes + 255) & ~(size_t)255;
        return r;
    };
    size_t NB = (size_t)N * HID * sizeof(float);
    float*    m     = (float*)alloc(NB);
    float*    agg   = (float*)alloc(NB);
    float*    xres  = (float*)alloc(NB);
    float*    nrm   = (float*)alloc((size_t)E * sizeof(float));
    float*    deg   = (float*)alloc((size_t)N * sizeof(float));
    float*    dinv  = (float*)alloc((size_t)N * sizeof(float));
    float*    dinv2 = (float*)alloc((size_t)N * sizeof(float));
    _Float16* wp0   = (_Float16*)alloc((size_t)8 * 8 * 32 * 16 * 2);
    _Float16* wpr   = (_Float16*)alloc((size_t)8 * 8 * 32 * 16 * 2);
    _Float16* wp1   = (_Float16*)alloc((size_t)4 * 8 * 32 * 16 * 2);
    _Float16* wp2   = (_Float16*)alloc((size_t)4 * 8 * 32 * 16 * 2);

    const int T = 256;
    // weight prepack (f32 -> f16 WMMA fragments)
    pack_w_kernel<<<(8 * 8 * 32 + T - 1) / T, T, 0, stream>>>(W0,   wp0, 256);
    pack_w_kernel<<<(8 * 8 * 32 + T - 1) / T, T, 0, stream>>>(resW, wpr, 256);
    pack_w_kernel<<<(4 * 8 * 32 + T - 1) / T, T, 0, stream>>>(W1,   wp1, 128);
    pack_w_kernel<<<(4 * 8 * 32 + T - 1) / T, T, 0, stream>>>(W2,   wp2, 128);

    // degree / norm
    deg_init_kernel<<<(N + T - 1) / T, T, 0, stream>>>(deg, N);
    deg_edge_kernel<<<(E + T - 1) / T, T, 0, stream>>>(dst, deg, E);
    dinv_kernel    <<<(N + T - 1) / T, T, 0, stream>>>(deg, dinv, dinv2, N);
    norm_kernel    <<<(E + T - 1) / T, T, 0, stream>>>(src, dst, dinv, nrm, E);

    dim3 gGemm((N + 15) / 16);
    dim3 gNode32(((size_t)N * 32 + T - 1) / T);
    dim3 gEdge((E + 7) / 8);
    dim3 gWaveNode((N + 7) / 8);

    // ---- layer 0 ----
    gemm_wmma_kernel<<<gGemm, T, 0, stream>>>(x, wp0, m,    N, 256);
    gemm_wmma_kernel<<<gGemm, T, 0, stream>>>(x, wpr, xres, N, 256);
    agg_init_kernel   <<<gNode32,   T, 0, stream>>>(m, b0, dinv2, agg, N);
    edge_scatter_kernel<<<gEdge,    T, 0, stream>>>(src, dst, nrm, m, agg, E);
    ln_relu_res_kernel<<<gWaveNode, T, 0, stream>>>(agg, g0, be0, xres, h, N);

    // ---- layer 1 ----
    gemm_wmma_kernel<<<gGemm, T, 0, stream>>>(h, wp1, m, N, 128);
    agg_init_kernel   <<<gNode32,   T, 0, stream>>>(m, b1, dinv2, agg, N);
    edge_scatter_kernel<<<gEdge,    T, 0, stream>>>(src, dst, nrm, m, agg, E);
    ln_relu_res_kernel<<<gWaveNode, T, 0, stream>>>(agg, g1, be1, h, h, N);

    // ---- layer 2 ----
    gemm_wmma_kernel<<<gGemm, T, 0, stream>>>(h, wp2, m, N, 128);
    agg_init_kernel   <<<gNode32,   T, 0, stream>>>(m, b2, dinv2, agg, N);
    edge_scatter_kernel<<<gEdge,    T, 0, stream>>>(src, dst, nrm, m, agg, E);
    ln_relu_res_kernel<<<gWaveNode, T, 0, stream>>>(agg, g2, be2, h, h, N);
}
